// Connect4GAT_14207751815180
// MI455X (gfx1250) — matrix operand order repaired and checked by reading, not compiled, not run on hardware
//
#include <hip/hip_runtime.h>
#include <hip/hip_bf16.h>
#include <stdint.h>

// ---------------------------------------------------------------------------
// Connect4 GAT forward, CDNA5 (gfx1250, wave32, WMMA bf16 + TDM async tile DMA)
// ---------------------------------------------------------------------------

#define BXW 7
#define BYH 6
#define NNODE 42
#define NHEAD 4
#define OUTF 128
#define HID 512
#define NB 1024
#define NTOT (NB * NNODE)       // 43008
#define NEDGE_L 262
#define NEDGE (NB * NEDGE_L)    // 268288
#define FC_IN (HID * NNODE)     // 21504

typedef __bf16 bf16_t;
typedef __attribute__((ext_vector_type(16))) __bf16 v16bf;
typedef __attribute__((ext_vector_type(8)))  __bf16 v8bf;
typedef __attribute__((ext_vector_type(8)))  float  v8f;
typedef __attribute__((ext_vector_type(4)))  unsigned int v4u;
typedef __attribute__((ext_vector_type(8)))  int v8i;
typedef __attribute__((ext_vector_type(4)))  int v4i;

union V16 { v16bf v; v8bf h[2]; };

// ---- fixed Connect4 graph topology (identical per board), baked at compile time
struct Graph {
    short src[NEDGE_L], tgt[NEDGE_L];
    short in_cnt[NNODE];
    short in_src[NNODE][8];
};

constexpr Graph make_graph() {
    Graph g{};
    int e = 0;
    for (int r = 0; r < BYH; ++r)
        for (int c = 0; c < BXW - 1; ++c) {
            int a = r * BXW + c, b = a + 1;
            g.src[e] = (short)a; g.tgt[e] = (short)b; ++e;
            g.src[e] = (short)b; g.tgt[e] = (short)a; ++e;
        }
    for (int r = 0; r < BYH - 1; ++r)
        for (int c = 0; c < BXW; ++c) {
            int a = r * BXW + c, b = a + BXW;
            g.src[e] = (short)a; g.tgt[e] = (short)b; ++e;
            g.src[e] = (short)b; g.tgt[e] = (short)a; ++e;
        }
    for (int r = 0; r < BYH - 1; ++r)
        for (int c = 0; c < BXW - 1; ++c) {
            int a = r * BXW + c, b = a + BXW + 1;
            g.src[e] = (short)a; g.tgt[e] = (short)b; ++e;
            g.src[e] = (short)b; g.tgt[e] = (short)a; ++e;
        }
    for (int r = 0; r < BYH - 1; ++r)
        for (int c = 1; c < BXW; ++c) {
            int a = r * BXW + c, b = a + BXW - 1;
            g.src[e] = (short)a; g.tgt[e] = (short)b; ++e;
            g.src[e] = (short)b; g.tgt[e] = (short)a; ++e;
        }
    for (int n = 0; n < NNODE; ++n) g.in_cnt[n] = 0;
    for (int i = 0; i < NEDGE_L; ++i) {
        int t = g.tgt[i];
        g.in_src[t][g.in_cnt[t]] = g.src[i];
        g.in_cnt[t] = (short)(g.in_cnt[t] + 1);
    }
    return g;
}

__constant__ Graph G = make_graph();

// ---- helpers
__device__ __forceinline__ int cl(float v) { return v == 0.f ? 0 : (v > 0.f ? 1 : 2); }
__device__ __forceinline__ float lrelu(float x) { return x > 0.f ? x : 0.2f * x; }
__device__ __forceinline__ float elu1(float x)  { return x > 0.f ? x : (__expf(x) - 1.f); }
// order-preserving float<->int encoding for atomicMax on possibly-negative floats
__device__ __forceinline__ int   fenc(float f) { int i = __float_as_int(f); return i >= 0 ? i : (i ^ 0x7fffffff); }
__device__ __forceinline__ float fdec(int i)   { return __int_as_float(i >= 0 ? i : (i ^ 0x7fffffff)); }

// ---------------------------------------------------------------------------
// f32 -> bf16 conversion
// ---------------------------------------------------------------------------
__global__ __launch_bounds__(256) void k_cvt(const float* __restrict__ in,
                                             bf16_t* __restrict__ out, int n) {
    int i = blockIdx.x * 256 + threadIdx.x;
    if (i < n) out[i] = (bf16_t)in[i];
}

// ---------------------------------------------------------------------------
// GAT1 precompute: ss/st per (class, head) — proj1 = W1[class] since x one-hot.
// misc[0..11]=ssM, misc[12..23]=stM, (int)misc[24]=gmax1, (int)misc[25]=gmax2
// ---------------------------------------------------------------------------
__global__ __launch_bounds__(384)
void k_init(const float* __restrict__ W1, const float* __restrict__ as1,
            const float* __restrict__ at1, float* __restrict__ misc) {
    int tid = threadIdx.x;
    if (tid < 2) ((int*)(misc + 24))[tid] = (int)0x80000000;  // -inf encoded
    int wv = tid >> 5, lane = tid & 31;
    int cls_ = wv >> 2, h = wv & 3;  // 12 waves = 3 classes x 4 heads
    float ss = 0.f, st = 0.f;
    for (int i = lane; i < OUTF; i += 32) {
        float v = W1[cls_ * HID + h * OUTF + i];
        ss += v * as1[h * OUTF + i];
        st += v * at1[h * OUTF + i];
    }
    for (int off = 16; off; off >>= 1) {
        ss += __shfl_xor(ss, off);
        st += __shfl_xor(st, off);
    }
    if (lane == 0) { misc[cls_ * 4 + h] = ss; misc[12 + cls_ * 4 + h] = st; }
}

// ---------------------------------------------------------------------------
// GAT1 global score max over all edges x heads (reference: scores - max(scores))
// ---------------------------------------------------------------------------
__global__ __launch_bounds__(256)
void k_scores1(const float* __restrict__ s, const float* __restrict__ misc,
               int* __restrict__ gmax) {
    int e = blockIdx.x * 256 + threadIdx.x;      // grid sized exactly NEDGE
    int b = e / NEDGE_L, el = e - b * NEDGE_L;
    int cs = cl(s[b * NNODE + G.src[el]]);
    int ct = cl(s[b * NNODE + G.tgt[el]]);
    const float* ssM = misc;
    const float* stM = misc + 12;
    float m = -3.4e38f;
    #pragma unroll
    for (int h = 0; h < NHEAD; ++h)
        m = fmaxf(m, lrelu(ssM[cs * 4 + h] + stM[ct * 4 + h]));
    int im = fenc(m);
    for (int off = 16; off; off >>= 1) im = max(im, __shfl_xor(im, off));
    if ((threadIdx.x & 31) == 0) atomicMax(gmax, im);
}

// ---------------------------------------------------------------------------
// GAT1 aggregate (gather form, no atomics): one block per node
// ---------------------------------------------------------------------------
__global__ __launch_bounds__(256)
void k_agg1(const float* __restrict__ s, const float* __restrict__ W1,
            const float* __restrict__ skip1, const float* __restrict__ b1,
            const float* __restrict__ misc, bf16_t* __restrict__ h1) {
    int n = blockIdx.x;
    int b = n / NNODE, ln = n - b * NNODE;
    int ct = cl(s[n]);
    int ec = G.in_cnt[ln];
    int cs[8];
    #pragma unroll
    for (int j = 0; j < 8; ++j)
        cs[j] = (j < ec) ? cl(s[b * NNODE + G.in_src[ln][j]]) : 0;
    float gmax = fdec(((const int*)(misc + 24))[0]);
    const float* ssM = misc;
    const float* stM = misc + 12;
    for (int f = threadIdx.x; f < HID; f += 256) {
        int h = f >> 7;
        float sth = stM[ct * 4 + h];
        float denom = 1e-16f, acc = 0.f, ex[8];
        for (int j = 0; j < ec; ++j) {
            float sc = lrelu(ssM[cs[j] * 4 + h] + sth);
            ex[j] = __expf(sc - gmax);
            denom += ex[j];
        }
        for (int j = 0; j < ec; ++j) acc += ex[j] * W1[cs[j] * HID + f];
        float o = acc / denom + skip1[ct * HID + f] + b1[f];
        o = elu1(elu1(o));
        h1[(size_t)n * HID + f] = (bf16_t)o;
    }
}

// ---------------------------------------------------------------------------
// Tiled bf16 WMMA GEMM with TDM A-tile staging.
// Block tile 128x64, 256 threads = 8 wave32 waves (4 in M, 2 in N),
// each wave owns 2x2 v_wmma_f32_16x16x32_bf16 accumulators.
// A tile (128x32 bf16) is DMA'd Global->LDS by the Tensor Data Mover with
// hardware row padding (16 DWORD rows + 8 DWORD pad -> 96B stride = [128][48]).
// Requires M%128==0, N%64==0, K%32==0 (true for every call here).
// ---------------------------------------------------------------------------
__global__ __launch_bounds__(256)
void k_gemm_bf16(const bf16_t* __restrict__ A, const bf16_t* __restrict__ B,
                 float* __restrict__ C, bf16_t* __restrict__ Cb,
                 const float* __restrict__ bias,
                 int M, int N, int K, int act) {
    __shared__ __align__(16) bf16_t As[128][48];  // row-major  [m][k], TDM-padded
    __shared__ __align__(16) bf16_t Bs[64][48];   // transposed [n][k], padded

    const int tid  = threadIdx.x;
    const int lane = tid & 31, wave = tid >> 5;
    const int wm = wave & 3, wn = wave >> 2;
    const int hs = lane >> 4, lm = lane & 15;   // half-select / row-in-half
    const int n0 = blockIdx.x * 64;
    const int m0 = blockIdx.y * 128;

    v8f acc[2][2] = {};

    const int bkr = tid >> 3, bnc = (tid & 7) << 3;   // B: 32 k x 64 n

    // --- invariant parts of the Tensor DMA descriptor (all uniform -> SGPRs)
    const unsigned lds_base = (unsigned)(size_t)(&As[0][0]);  // LDS byte offset
    const unsigned long long gbase =
        (unsigned long long)(size_t)A + (unsigned long long)m0 * (unsigned)K * 2ull;
    v8i g1;
    g1[0] = (1 << 16)            // data_size = 1 (2 bytes, bf16)
          | (1 << 20)            // pad_enable
          | (3 << 22)            // pad_interval: 16 DWORDs (one 64B tile row)
          | (7 << 25);           // pad_amount:   8 DWORDs (32B) -> 96B row stride
    g1[1] = (int)(((unsigned)K & 0xFFFFu) << 16);              // tensor_dim0 lo
    g1[2] = (int)(((unsigned)K >> 16) | (((unsigned)M & 0xFFFFu) << 16));
    g1[3] = (int)(((unsigned)M >> 16) | (32u << 16));          // tile_dim0 = 32
    g1[4] = 128;                                               // tile_dim1 = 128
    g1[5] = (int)(unsigned)K;                                  // dim0 stride lo
    g1[6] = 0;                                                 // stride hi / dim1 stride
    g1[7] = 0;
    const v4i gz  = {0, 0, 0, 0};                              // 2D tile: groups 2/3 unused
    const v8i gz8 = {0, 0, 0, 0, 0, 0, 0, 0};                  // upper-dim group unused

    for (int k0 = 0; k0 < K; k0 += 32) {
        // ---- A tile: Tensor Data Mover DMA (one issue per workgroup, wave 0)
        if (wave == 0) {
            unsigned long long ga = gbase + (unsigned long long)k0 * 2ull;
            v4u g0;
            g0[0] = 1u;                         // count=1, user descriptor
            g0[1] = lds_base;                   // LDS destination
            g0[2] = (unsigned)ga;               // global_addr[31:0]
            g0[3] = (unsigned)((ga >> 32) & 0x01FFFFFFu) | (2u << 30);  // addr hi | type=2
            __builtin_amdgcn_tensor_load_to_lds(g0, g1, gz, gz, gz8, 0);
        }
        // ---- B tile transposed (coalesced global read, scattered LDS write)
        const bf16_t* gB = B + (size_t)(k0 + bkr) * N + (n0 + bnc);
        v8bf bv = *(const v8bf*)gB;
        #pragma unroll
        for (int i = 0; i < 8; ++i) Bs[bnc + i][bkr] = bv[i];

        if (k0 + 32 < K)  // global_prefetch_b8 for next B tile
            __builtin_prefetch(B + (size_t)(k0 + 32 + bkr) * N + (n0 + bnc), 0, 1);

        if (wave == 0) __builtin_amdgcn_s_wait_tensorcnt(0);  // TDM done before publish
        __syncthreads();

        // A fragment per ISA 16-bit 16x32 layout:
        //   lane<16: K 0..7 & 16..23 ; lane>=16: K 8..15 & 24..31
        V16 a[2], bfr[2];
        #pragma unroll
        for (int tm = 0; tm < 2; ++tm) {
            const bf16_t* ap = &As[wm * 32 + tm * 16 + lm][0];
            a[tm].h[0] = *(const v8bf*)(ap + hs * 8);
            a[tm].h[1] = *(const v8bf*)(ap + hs * 8 + 16);
        }
        // B fragment: lane<16 holds K 0..7 & 16..23 of column n, lane>=16 holds K 8..15 & 24..31
        #pragma unroll
        for (int tn = 0; tn < 2; ++tn) {
            const bf16_t* bp = &Bs[wn * 32 + tn * 16 + lm][0];
            bfr[tn].h[0] = *(const v8bf*)(bp + hs * 8);
            bfr[tn].h[1] = *(const v8bf*)(bp + hs * 8 + 16);
        }
        #pragma unroll
        for (int tm = 0; tm < 2; ++tm)
            #pragma unroll
            for (int tn = 0; tn < 2; ++tn)
                acc[tm][tn] = __builtin_amdgcn_wmma_f32_16x16x32_bf16(
                    false, a[tm].v, false, bfr[tn].v,
                    (short)0, acc[tm][tn], false, false);
        __syncthreads();
    }

    // epilogue: C/D layout — VGPR r holds row (hs*8 + r), col = lane&15
    #pragma unroll
    for (int tm = 0; tm < 2; ++tm)
        #pragma unroll
        for (int tn = 0; tn < 2; ++tn)
            #pragma unroll
            for (int r = 0; r < 8; ++r) {
                int m = m0 + wm * 32 + tm * 16 + hs * 8 + r;
                int n = n0 + wn * 32 + tn * 16 + lm;
                float v = acc[tm][tn][r];
                if (bias) v += bias[n];
                if (act == 1) v = v > 0.f ? v : 0.f;
                size_t off = (size_t)m * N + n;
                if (C)  C[off]  = v;
                if (Cb) Cb[off] = (bf16_t)v;
            }
}

// ---------------------------------------------------------------------------
// GAT2 per-node attention logits: ss2/st2[n][h] = <proj2[n,h,:], a_src/tgt[h]>
// ---------------------------------------------------------------------------
__global__ __launch_bounds__(128)
void k_ss2(const float* __restrict__ proj2, const float* __restrict__ as2,
           const float* __restrict__ at2, float* __restrict__ ss2,
           float* __restrict__ st2) {
    int n = blockIdx.x;
    int h = threadIdx.x >> 5, lane = threadIdx.x & 31;
    const float* p = proj2 + (size_t)n * HID + h * OUTF;
    float ss = 0.f, st = 0.f;
    for (int i = lane; i < OUTF; i += 32) {
        float v = p[i];
        ss += v * as2[h * OUTF + i];
        st += v * at2[h * OUTF + i];
    }
    for (int off = 16; off; off >>= 1) {
        ss += __shfl_xor(ss, off);
        st += __shfl_xor(st, off);
    }
    if (lane == 0) { ss2[n * 4 + h] = ss; st2[n * 4 + h] = st; }
}

__global__ __launch_bounds__(256)
void k_scores2(const float* __restrict__ ss2, const float* __restrict__ st2,
               int* __restrict__ gmax) {
    int e = blockIdx.x * 256 + threadIdx.x;
    int b = e / NEDGE_L, el = e - b * NEDGE_L;
    int sg = b * NNODE + G.src[el], tg = b * NNODE + G.tgt[el];
    float m = -3.4e38f;
    #pragma unroll
    for (int h = 0; h < NHEAD; ++h)
        m = fmaxf(m, lrelu(ss2[sg * 4 + h] + st2[tg * 4 + h]));
    int im = fenc(m);
    for (int off = 16; off; off >>= 1) im = max(im, __shfl_xor(im, off));
    if ((threadIdx.x & 31) == 0) atomicMax(gmax, im);
}

__global__ __launch_bounds__(256)
void k_agg2(const float* __restrict__ proj2, const float* __restrict__ skip2,
            const float* __restrict__ ss2, const float* __restrict__ st2,
            const float* __restrict__ b2, const float* __restrict__ misc,
            bf16_t* __restrict__ h2) {
    int n = blockIdx.x;
    int b = n / NNODE, ln = n - b * NNODE;
    int ec = G.in_cnt[ln];
    int sg[8];
    #pragma unroll
    for (int j = 0; j < 8; ++j)
        sg[j] = (j < ec) ? (b * NNODE + G.in_src[ln][j]) : n;
    float gmax = fdec(((const int*)(misc + 24))[1]);
    for (int f = threadIdx.x; f < HID; f += 256) {
        int h = f >> 7;
        float sth = st2[n * 4 + h];
        float denom = 1e-16f, acc = 0.f, ex[8];
        for (int j = 0; j < ec; ++j) {
            float sc = lrelu(ss2[sg[j] * 4 + h] + sth);
            ex[j] = __expf(sc - gmax);
            denom += ex[j];
        }
        for (int j = 0; j < ec; ++j)
            acc += ex[j] * proj2[(size_t)sg[j] * HID + f];
        float o = acc / denom + skip2[(size_t)n * HID + f] + b2[f];
        o = elu1(elu1(o));
        h2[(size_t)n * HID + f] = (bf16_t)o;
    }
}

// ---------------------------------------------------------------------------
// policy / value heads + log_softmax + tanh. one block (128 thr) per board.
// ---------------------------------------------------------------------------
__global__ __launch_bounds__(128)
void k_head(const float* __restrict__ h, const float* __restrict__ pW,
            const float* __restrict__ pb, const float* __restrict__ vW,
            const float* __restrict__ vb, float* __restrict__ out) {
    __shared__ float hs[128];
    __shared__ float logits[8];
    int b = blockIdx.x, tid = threadIdx.x;
    hs[tid] = h[(size_t)b * 128 + tid];
    __syncthreads();
    if (tid < 7) {
        float a = pb[tid];
        for (int i = 0; i < 128; ++i) a += hs[i] * pW[i * 7 + tid];
        logits[tid] = a;
    } else if (tid == 7) {
        float a = vb[0];
        for (int i = 0; i < 128; ++i) a += hs[i] * vW[i];
        logits[7] = a;
    }
    __syncthreads();
    if (tid == 0) {
        float mx = logits[0];
        #pragma unroll
        for (int j = 1; j < 7; ++j) mx = fmaxf(mx, logits[j]);
        float se = 0.f;
        #pragma unroll
        for (int j = 0; j < 7; ++j) se += __expf(logits[j] - mx);
        float lse = mx + __logf(se);
        #pragma unroll
        for (int j = 0; j < 7; ++j) out[b * 7 + j] = logits[j] - lse;
        out[NB * 7 + b] = tanhf(logits[7]);
    }
}

// ---------------------------------------------------------------------------
extern "C" void kernel_launch(void* const* d_in, const int* in_sizes, int n_in,
                              void* d_out, int out_size, void* d_ws, size_t ws_size,
                              hipStream_t stream) {
    const float* s        = (const float*)d_in[0];
    // d_in[1] = edge_index (fixed pattern, baked into __constant__ Graph)
    const float* gat1_W    = (const float*)d_in[2];
    const float* gat1_asrc = (const float*)d_in[3];
    const float* gat1_atgt = (const float*)d_in[4];
    const float* gat1_skip = (const float*)d_in[5];
    const float* gat1_b    = (const float*)d_in[6];
    const float* gat2_W    = (const float*)d_in[7];
    const float* gat2_asrc = (const float*)d_in[8];
    const float* gat2_atgt = (const float*)d_in[9];
    const float* gat2_skip = (const float*)d_in[10];
    const float* gat2_b    = (const float*)d_in[11];
    const float* fc1_W = (const float*)d_in[12];
    const float* fc1_b = (const float*)d_in[13];
    const float* fc2_W = (const float*)d_in[14];
    const float* fc2_b = (const float*)d_in[15];
    const float* pol_W = (const float*)d_in[16];
    const float* pol_b = (const float*)d_in[17];
    const float* val_W = (const float*)d_in[18];
    const float* val_b = (const float*)d_in[19];
    float* out = (float*)d_out;

    // bump allocator over workspace
    char* w = (char*)d_ws;
    auto alloc = [&](size_t bytes) -> void* {
        void* p = (void*)w;
        w += (bytes + 255) & ~(size_t)255;
        return p;
    };
    bf16_t* h1     = (bf16_t*)alloc((size_t)NTOT * HID * 2);
    bf16_t* h2     = (bf16_t*)alloc((size_t)NTOT * HID * 2);
    float*  proj2  = (float*) alloc((size_t)NTOT * HID * 4);
    float*  skip2o = (float*) alloc((size_t)NTOT * HID * 4);
    float*  ss2    = (float*) alloc((size_t)NTOT * 4 * 4);
    float*  st2    = (float*) alloc((size_t)NTOT * 4 * 4);
    bf16_t* w2bf   = (bf16_t*)alloc((size_t)HID * HID * 2);
    bf16_t* sk2bf  = (bf16_t*)alloc((size_t)HID * HID * 2);
    bf16_t* fc1wbf = (bf16_t*)alloc((size_t)FC_IN * 256 * 2);
    bf16_t* fc2wbf = (bf16_t*)alloc((size_t)256 * 128 * 2);
    bf16_t* fc1obf = (bf16_t*)alloc((size_t)NB * 256 * 2);
    float*  fc2o   = (float*) alloc((size_t)NB * 128 * 4);
    float*  misc   = (float*) alloc(64 * 4);
    int* gmax1 = (int*)(misc + 24);
    int* gmax2 = (int*)(misc + 25);

    // weight conversions to bf16
    k_cvt<<<(HID * HID + 255) / 256, 256, 0, stream>>>(gat2_W, w2bf, HID * HID);
    k_cvt<<<(HID * HID + 255) / 256, 256, 0, stream>>>(gat2_skip, sk2bf, HID * HID);
    k_cvt<<<(FC_IN * 256 + 255) / 256, 256, 0, stream>>>(fc1_W, fc1wbf, FC_IN * 256);
    k_cvt<<<(256 * 128 + 255) / 256, 256, 0, stream>>>(fc2_W, fc2wbf, 256 * 128);

    // GAT layer 1 (one-hot input -> pure gather path)
    k_init<<<1, 384, 0, stream>>>(gat1_W, gat1_asrc, gat1_atgt, misc);
    k_scores1<<<NEDGE / 256, 256, 0, stream>>>(s, misc, gmax1);
    k_agg1<<<NTOT, 256, 0, stream>>>(s, gat1_W, gat1_skip, gat1_b, misc, h1);

    // GAT layer 2 projections (bf16 WMMA GEMMs, TDM-staged A tiles)
    k_gemm_bf16<<<dim3(HID / 64, NTOT / 128), 256, 0, stream>>>(
        h1, w2bf, proj2, nullptr, nullptr, NTOT, HID, HID, 0);
    k_gemm_bf16<<<dim3(HID / 64, NTOT / 128), 256, 0, stream>>>(
        h1, sk2bf, skip2o, nullptr, nullptr, NTOT, HID, HID, 0);

    // GAT layer 2 attention + aggregation
    k_ss2<<<NTOT, 128, 0, stream>>>(proj2, gat2_asrc, gat2_atgt, ss2, st2);
    k_scores2<<<NEDGE / 256, 256, 0, stream>>>(ss2, st2, gmax2);
    k_agg2<<<NTOT, 256, 0, stream>>>(proj2, skip2o, ss2, st2, gat2_b, misc, h2);

    // FC stack: h2 viewed as [1024, 21504] (contiguous)
    k_gemm_bf16<<<dim3(256 / 64, NB / 128), 256, 0, stream>>>(
        h2, fc1wbf, nullptr, fc1obf, fc1_b, NB, 256, FC_IN, 1);
    k_gemm_bf16<<<dim3(128 / 64, NB / 128), 256, 0, stream>>>(
        fc1obf, fc2wbf, fc2o, nullptr, fc2_b, NB, 128, 256, 1);

    // heads: log_softmax policy + tanh value
    k_head<<<NB, 128, 0, stream>>>(fc2o, pol_W, pol_b, val_W, val_b, out);

    (void)in_sizes; (void)n_in; (void)out_size; (void)ws_size;
}
